// ReactantProductAttention_592705487198
// MI455X (gfx1250) — matrix operand order repaired and checked by reading, not compiled, not run on hardware
//
#include <hip/hip_runtime.h>
#include <hip/hip_bf16.h>

typedef __attribute__((ext_vector_type(16))) __bf16 bf16x16;
typedef __attribute__((ext_vector_type(8)))  float  f32x8;

constexpr int HD  = 768;     // hidden dim
constexpr int NB  = 512;     // num graphs
constexpr int LM  = 128;     // max dense len
constexpr int BD  = NB * LM; // dense rows = 65536
constexpr int NHh = 8;       // heads
constexpr int DH  = 96;      // head dim

// ---------------------------------------------------------------------------
// helpers
// ---------------------------------------------------------------------------
static __device__ inline bf16x16 ld_frag2(const __bf16* p0, const __bf16* p1) {
  union U { bf16x16 v; uint4 u[2]; } r;
  r.u[0] = *(const uint4*)p0;
  r.u[1] = *(const uint4*)p1;
  return r.v;
}

static __device__ inline f32x8 wmma_bf16(bf16x16 a, bf16x16 b, f32x8 c) {
  return __builtin_amdgcn_wmma_f32_16x16x32_bf16(
      /*neg_a=*/false, a, /*neg_b=*/false, b,
      /*c_mod=*/(short)0, c, /*reuse_a=*/false, /*reuse_b=*/false);
}

// ---------------------------------------------------------------------------
// prep kernels
// ---------------------------------------------------------------------------
__global__ void f32_to_bf16_kernel(const float* __restrict__ s,
                                   __bf16* __restrict__ d, int n) {
  int i = blockIdx.x * blockDim.x + threadIdx.x;
  int stride = gridDim.x * blockDim.x;
  for (; i < n; i += stride) d[i] = (__bf16)s[i];
}

__global__ void graph_ranges_kernel(const int* __restrict__ batch, int Nn,
                                    int* __restrict__ start,
                                    int* __restrict__ count) {
  int i = blockIdx.x * blockDim.x + threadIdx.x;
  if (i < Nn) {
    int g = batch[i];
    atomicAdd(&count[g], 1);
    atomicMin(&start[g], i);
  }
}

// one thread per graph: serial position assignment within the graph
__global__ void positions_kernel(const int* __restrict__ aot,
                                 const int* __restrict__ start,
                                 const int* __restrict__ count,
                                 int* __restrict__ dense_row,
                                 int* __restrict__ nid_r,
                                 int* __restrict__ nid_p,
                                 int* __restrict__ nr,
                                 int* __restrict__ np_) {
  int g = blockIdx.x * blockDim.x + threadIdx.x;
  if (g >= NB) return;
  int s = start[g], c = count[g];
  int ir = 0, ip = 0;
  for (int t = 0; t < c; ++t) {
    int node = s + t;
    if (aot[node] == 0) {
      int row = g * LM + ir;
      dense_row[node] = row;          // reactant side
      nid_r[row] = node;
      ++ir;
    } else {
      int row = g * LM + ip;
      dense_row[node] = BD + row;     // product side (offset flag)
      nid_p[row] = node;
      ++ip;
    }
  }
  nr[g] = ir;
  np_[g] = ip;
}

// one block per node: copy 768 floats -> bf16 dense slot
__global__ void scatter_kernel(const float* __restrict__ nf,
                               const int* __restrict__ dense_row,
                               __bf16* __restrict__ rd,
                               __bf16* __restrict__ pd) {
  int node = blockIdx.x;
  int dr = dense_row[node];
  __bf16* dst = (dr < BD) ? (rd + (size_t)dr * HD)
                          : (pd + (size_t)(dr - BD) * HD);
  const float* src = nf + (size_t)node * HD;
  for (int c = threadIdx.x; c < HD; c += blockDim.x) dst[c] = (__bf16)src[c];
}

// ---------------------------------------------------------------------------
// WMMA GEMM:  Y[M,N] = A[M,K](bf16) * Bw[N,K]^T(bf16) + bias
// block: 256 thr = 8 waves; block tile 128(M) x 128(N);
// wave grid 4(M) x 2(N), wave tile 32 x 64; K-step 32.
// Software-pipelined: next K-tile's global loads issued before the WMMA clause.
// MODE: 0 = outb bf16
//       1 = gather residual: att[node] = nf[node] + y   (outf f32 + outb bf16)
//       2 = relu -> outb bf16
//       3 = outf f32 = resid + y
// ---------------------------------------------------------------------------
template <int MODE>
__global__ __launch_bounds__(256) void gemm_bf16_kernel(
    const __bf16* __restrict__ A, const __bf16* __restrict__ Bw,
    const float* __restrict__ bias,
    int M, int K, int N,
    __bf16* __restrict__ outb, float* __restrict__ outf,
    const float* __restrict__ resid,
    const float* __restrict__ nf, const int* __restrict__ node_id) {
  __shared__ alignas(16) __bf16 As[128 * 32];
  __shared__ alignas(16) __bf16 Bs[128 * 32];

  const int t = threadIdx.x;
  const int lane = t & 31, wid = t >> 5;
  const int ln = lane & 15, hi = lane >> 4;
  const int waveM = wid >> 1, waveN = wid & 1;
  const int mBlk = blockIdx.x * 128, nBlk = blockIdx.y * 128;

  // global->LDS staging: 128x32 bf16 per tile = 512 chunks of 8 bf16; 2/thread
  const int r0 = t >> 2, c0 = (t & 3) * 8;
  const int r1 = r0 + 64;
  const bool a0ok = (mBlk + r0) < M;
  const bool a1ok = (mBlk + r1) < M;
  const __bf16* a0p = A + (size_t)(mBlk + r0) * K + c0;
  const __bf16* a1p = A + (size_t)(mBlk + r1) * K + c0;
  const __bf16* b0p = Bw + (size_t)(nBlk + r0) * K + c0;
  const __bf16* b1p = Bw + (size_t)(nBlk + r1) * K + c0;

  f32x8 acc[2][4] = {};

  const uint4 zero4 = {0u, 0u, 0u, 0u};
  uint4 a0 = a0ok ? *(const uint4*)a0p : zero4;
  uint4 a1 = a1ok ? *(const uint4*)a1p : zero4;
  uint4 b0 = *(const uint4*)b0p;
  uint4 b1 = *(const uint4*)b1p;

  for (int k0 = 0; k0 < K; k0 += 32) {
    __syncthreads();
    *(uint4*)&As[r0 * 32 + c0] = a0;
    *(uint4*)&As[r1 * 32 + c0] = a1;
    *(uint4*)&Bs[r0 * 32 + c0] = b0;
    *(uint4*)&Bs[r1 * 32 + c0] = b1;
    __syncthreads();

    // prefetch next K-tile into registers (overlaps with WMMA clause below)
    const int kn = k0 + 32;
    if (kn < K) {
      a0 = a0ok ? *(const uint4*)(a0p + kn) : zero4;
      a1 = a1ok ? *(const uint4*)(a1p + kn) : zero4;
      b0 = *(const uint4*)(b0p + kn);
      b1 = *(const uint4*)(b1p + kn);
      if (a0ok && kn + 32 < K) __builtin_prefetch(a0p + kn + 32, 0, 1);
    }

    bf16x16 af[2], bfr[4];
#pragma unroll
    for (int i = 0; i < 2; ++i) {
      const int r = waveM * 32 + i * 16 + ln;
      af[i] = ld_frag2(&As[r * 32 + hi * 8], &As[r * 32 + 16 + hi * 8]);
    }
#pragma unroll
    for (int j = 0; j < 4; ++j) {
      const int r = waveN * 64 + j * 16 + ln;
      bfr[j] = ld_frag2(&Bs[r * 32 + hi * 8], &Bs[r * 32 + 16 + hi * 8]);
    }
#pragma unroll
    for (int i = 0; i < 2; ++i)
#pragma unroll
      for (int j = 0; j < 4; ++j) acc[i][j] = wmma_bf16(af[i], bfr[j], acc[i][j]);
  }

  // epilogue: D layout -> row = r + 8*hi, col = lane&15 within each 16x16 tile
#pragma unroll
  for (int j = 0; j < 4; ++j) {
    const int gcol = nBlk + waveN * 64 + j * 16 + ln;
    const float bv = bias[gcol];
#pragma unroll
    for (int i = 0; i < 2; ++i) {
#pragma unroll
      for (int r = 0; r < 8; ++r) {
        const int grow = mBlk + waveM * 32 + i * 16 + r + 8 * hi;
        if (grow >= M) continue;
        float v = acc[i][j][r] + bv;
        if (MODE == 0) {
          outb[(size_t)grow * N + gcol] = (__bf16)v;
        } else if (MODE == 1) {
          const int nd = node_id[grow];
          if (nd >= 0) {
            const float o = nf[(size_t)nd * N + gcol] + v;
            outf[(size_t)nd * N + gcol] = o;
            outb[(size_t)nd * N + gcol] = (__bf16)o;
          }
        } else if (MODE == 2) {
          v = fmaxf(v, 0.0f);
          outb[(size_t)grow * N + gcol] = (__bf16)v;
        } else {
          outf[(size_t)grow * N + gcol] = resid[(size_t)grow * N + gcol] + v;
        }
      }
    }
  }
}

// ---------------------------------------------------------------------------
// cross-attention for one (graph, head): O = softmax(mask(Q K^T / sqrt(Dh))) V
// block = 256 thr = 8 waves; wave w owns query rows [16w, 16w+16)
// ---------------------------------------------------------------------------
__global__ __launch_bounds__(256) void attn_kernel(
    const __bf16* __restrict__ Q, const __bf16* __restrict__ Kt,
    const __bf16* __restrict__ V, const int* __restrict__ kvlen_arr,
    __bf16* __restrict__ O) {
  __shared__ alignas(16) __bf16 Qs[LM * DH];      // 24 KB
  __shared__ alignas(16) __bf16 Ks[LM * DH];      // 24 KB
  __shared__ alignas(16) __bf16 Vt[DH * LM];      // 24 KB, transposed
  __shared__ alignas(16) __bf16 Ps[8 * 16 * LM];  // 32 KB, per-wave P tiles

  const int b = blockIdx.x, h = blockIdx.y;
  const int t = threadIdx.x;
  const size_t base = (size_t)b * LM * HD + (size_t)h * DH;

  // stage Q, K (row-major [128 x 96])
  for (int c = t; c < LM * (DH / 8); c += 256) {
    const int row = c / (DH / 8), col = (c % (DH / 8)) * 8;
    *(uint4*)&Qs[row * DH + col] = *(const uint4*)(Q + base + (size_t)row * HD + col);
    *(uint4*)&Ks[row * DH + col] = *(const uint4*)(Kt + base + (size_t)row * HD + col);
  }
  // stage V transposed -> [96 x 128]
  for (int e = t; e < LM * DH; e += 256) {
    const int row = e / DH, d = e % DH;
    Vt[d * LM + row] = V[base + (size_t)row * HD + d];
  }
  __syncthreads();

  const int lane = t & 31, wid = t >> 5;
  const int ln = lane & 15, hi = lane >> 4;
  const int kvlen = kvlen_arr[b];
  const float scale = 0.10206207261596575f;  // 1/sqrt(96)

  // S = Q K^T : 8 key tiles, 3 k-steps of 32 over Dh
  f32x8 S[8] = {};
  const __bf16* qrow = &Qs[(wid * 16 + ln) * DH];
  for (int ks = 0; ks < DH; ks += 32) {
    const bf16x16 a = ld_frag2(qrow + ks + hi * 8, qrow + ks + 16 + hi * 8);
#pragma unroll
    for (int j = 0; j < 8; ++j) {
      const __bf16* krow = &Ks[(j * 16 + ln) * DH + ks];
      const bf16x16 bm = ld_frag2(krow + hi * 8, krow + 16 + hi * 8);
      S[j] = wmma_bf16(a, bm, S[j]);
    }
  }

  // masked softmax over key dim: element (j,r,lane) = (row r+8*hi, key 16j+ln)
  float mx[8], sm[8];
#pragma unroll
  for (int r = 0; r < 8; ++r) { mx[r] = -3.0e38f; sm[r] = 0.0f; }
#pragma unroll
  for (int j = 0; j < 8; ++j) {
    const bool valid = (j * 16 + ln) < kvlen;
#pragma unroll
    for (int r = 0; r < 8; ++r) {
      const float s = valid ? S[j][r] * scale : -1.0e30f;
      S[j][r] = s;
      mx[r] = fmaxf(mx[r], s);
    }
  }
#pragma unroll
  for (int r = 0; r < 8; ++r)
    for (int off = 1; off < 16; off <<= 1)
      mx[r] = fmaxf(mx[r], __shfl_xor(mx[r], off, 32));
#pragma unroll
  for (int j = 0; j < 8; ++j)
#pragma unroll
    for (int r = 0; r < 8; ++r) {
      const float p = __expf(S[j][r] - mx[r]);
      S[j][r] = p;
      sm[r] += p;
    }
#pragma unroll
  for (int r = 0; r < 8; ++r) {
    for (int off = 1; off < 16; off <<= 1) sm[r] += __shfl_xor(sm[r], off, 32);
    sm[r] = 1.0f / sm[r];
  }

  // write P (bf16) to per-wave LDS tile [16 x 128] in A-matrix row-major form
  __bf16* prow = &Ps[wid * 16 * LM];
#pragma unroll
  for (int j = 0; j < 8; ++j)
#pragma unroll
    for (int r = 0; r < 8; ++r)
      prow[(r + 8 * hi) * LM + j * 16 + ln] = (__bf16)(S[j][r] * sm[r]);

  // O = P V : 6 d-tiles, 4 k-steps of 32 over keys
  f32x8 Oc[6] = {};
  const __bf16* parow = &Ps[wid * 16 * LM + ln * LM];
  for (int kk = 0; kk < LM; kk += 32) {
    const bf16x16 a = ld_frag2(parow + kk + hi * 8, parow + kk + 16 + hi * 8);
#pragma unroll
    for (int dt = 0; dt < 6; ++dt) {
      const __bf16* vrow = &Vt[(dt * 16 + ln) * LM + kk];
      const bf16x16 bm = ld_frag2(vrow + hi * 8, vrow + 16 + hi * 8);
      Oc[dt] = wmma_bf16(a, bm, Oc[dt]);
    }
  }

  __bf16* obase = O + (size_t)(b * LM + wid * 16) * HD + (size_t)h * DH;
#pragma unroll
  for (int dt = 0; dt < 6; ++dt)
#pragma unroll
    for (int r = 0; r < 8; ++r)
      obase[(size_t)(r + 8 * hi) * HD + dt * 16 + ln] = (__bf16)Oc[dt][r];
}

// ---------------------------------------------------------------------------
// host orchestration
// ---------------------------------------------------------------------------
extern "C" void kernel_launch(void* const* d_in, const int* in_sizes, int n_in,
                              void* d_out, int out_size, void* d_ws, size_t ws_size,
                              hipStream_t stream) {
  (void)n_in; (void)out_size; (void)ws_size;

  const float* nf     = (const float*)d_in[0];
  const float* Wqkv_r = (const float*)d_in[1];
  const float* bqkv_r = (const float*)d_in[2];
  const float* Wo_r   = (const float*)d_in[3];
  const float* bo_r   = (const float*)d_in[4];
  const float* Wqkv_p = (const float*)d_in[5];
  const float* bqkv_p = (const float*)d_in[6];
  const float* Wo_p   = (const float*)d_in[7];
  const float* bo_p   = (const float*)d_in[8];
  const float* W1     = (const float*)d_in[9];
  const float* b1     = (const float*)d_in[10];
  const float* W2     = (const float*)d_in[11];
  const float* b2     = (const float*)d_in[12];
  const int*   aot    = (const int*)d_in[13];
  const int*   batch  = (const int*)d_in[14];
  const int Nn = in_sizes[0] / HD;

  // workspace arena
  char* base = (char*)d_ws;
  size_t off = 0;
  auto alloc = [&](size_t bytes) -> void* {
    void* p = base + off;
    off += (bytes + 255) & ~(size_t)255;
    return p;
  };

  __bf16* wqkvr = (__bf16*)alloc((size_t)3 * HD * HD * 2);
  __bf16* wor   = (__bf16*)alloc((size_t)HD * HD * 2);
  __bf16* wqkvp = (__bf16*)alloc((size_t)3 * HD * HD * 2);
  __bf16* wop   = (__bf16*)alloc((size_t)HD * HD * 2);
  __bf16* w1b   = (__bf16*)alloc((size_t)2 * HD * HD * 2);
  __bf16* w2b   = (__bf16*)alloc((size_t)2 * HD * HD * 2);

  int* starti = (int*)alloc(NB * sizeof(int));
  int* counti = (int*)alloc(NB * sizeof(int));
  int* nri    = (int*)alloc(NB * sizeof(int));
  int* npi    = (int*)alloc(NB * sizeof(int));
  int* drow   = (int*)alloc((size_t)Nn * sizeof(int));
  int* nidr   = (int*)alloc((size_t)BD * sizeof(int));
  int* nidp   = (int*)alloc((size_t)BD * sizeof(int));

  const size_t DSZ = (size_t)BD * HD * 2;  // dense bf16 tensor bytes
  __bf16* rd = (__bf16*)alloc(DSZ);  // reactant dense; reused as attn_out_r
  __bf16* pd = (__bf16*)alloc(DSZ);  // product dense;  reused as attn_out_p
  __bf16* qr = (__bf16*)alloc(DSZ);
  __bf16* kr = (__bf16*)alloc(DSZ);
  __bf16* vr = (__bf16*)alloc(DSZ);
  __bf16* qp = (__bf16*)alloc(DSZ);
  __bf16* kp = (__bf16*)alloc(DSZ);
  __bf16* vp = (__bf16*)alloc(DSZ);

  float*  att  = (float*)alloc((size_t)Nn * HD * 4);
  __bf16* attb = (__bf16*)alloc((size_t)Nn * HD * 2);
  __bf16* h1   = (__bf16*)alloc((size_t)Nn * 2 * HD * 2);

  // weight conversion fp32 -> bf16
  auto cvt = [&](const float* s, __bf16* d, int n) {
    int blocks = (n + 255) / 256;
    if (blocks > 2048) blocks = 2048;
    f32_to_bf16_kernel<<<blocks, 256, 0, stream>>>(s, d, n);
  };
  cvt(Wqkv_r, wqkvr, 3 * HD * HD);
  cvt(Wo_r,   wor,   HD * HD);
  cvt(Wqkv_p, wqkvp, 3 * HD * HD);
  cvt(Wo_p,   wop,   HD * HD);
  cvt(W1,     w1b,   2 * HD * HD);
  cvt(W2,     w2b,   2 * HD * HD);

  // init
  hipMemsetAsync(starti, 0x7f, NB * sizeof(int), stream);
  hipMemsetAsync(counti, 0x00, NB * sizeof(int), stream);
  hipMemsetAsync(nidr, 0xff, (size_t)BD * sizeof(int), stream);
  hipMemsetAsync(nidp, 0xff, (size_t)BD * sizeof(int), stream);
  hipMemsetAsync(rd, 0x00, DSZ, stream);
  hipMemsetAsync(pd, 0x00, DSZ, stream);

  graph_ranges_kernel<<<(Nn + 255) / 256, 256, 0, stream>>>(batch, Nn, starti, counti);
  positions_kernel<<<(NB + 255) / 256, 256, 0, stream>>>(aot, starti, counti,
                                                         drow, nidr, nidp, nri, npi);
  scatter_kernel<<<Nn, 256, 0, stream>>>(nf, drow, rd, pd);

  auto gemm = [&](const __bf16* A, const __bf16* Bw, const float* bias,
                  int M, int K, int N, int mode,
                  __bf16* outb, float* outf, const float* resid, const int* nid) {
    dim3 g((M + 127) / 128, N / 128);
    switch (mode) {
      case 0:
        gemm_bf16_kernel<0><<<g, 256, 0, stream>>>(A, Bw, bias, M, K, N,
                                                   outb, outf, resid, nf, nid);
        break;
      case 1:
        gemm_bf16_kernel<1><<<g, 256, 0, stream>>>(A, Bw, bias, M, K, N,
                                                   outb, outf, resid, nf, nid);
        break;
      case 2:
        gemm_bf16_kernel<2><<<g, 256, 0, stream>>>(A, Bw, bias, M, K, N,
                                                   outb, outf, resid, nf, nid);
        break;
      default:
        gemm_bf16_kernel<3><<<g, 256, 0, stream>>>(A, Bw, bias, M, K, N,
                                                   outb, outf, resid, nf, nid);
        break;
    }
  };

  // QKV projections (mode 0, bf16 out)
  gemm(rd, wqkvr,                       bqkv_r,          BD, HD, HD, 0, qr, nullptr, nullptr, nullptr);
  gemm(pd, wqkvr + (size_t)HD * HD,     bqkv_r + HD,     BD, HD, HD, 0, kr, nullptr, nullptr, nullptr);
  gemm(pd, wqkvr + (size_t)2 * HD * HD, bqkv_r + 2 * HD, BD, HD, HD, 0, vr, nullptr, nullptr, nullptr);
  gemm(pd, wqkvp,                       bqkv_p,          BD, HD, HD, 0, qp, nullptr, nullptr, nullptr);
  gemm(rd, wqkvp + (size_t)HD * HD,     bqkv_p + HD,     BD, HD, HD, 0, kp, nullptr, nullptr, nullptr);
  gemm(rd, wqkvp + (size_t)2 * HD * HD, bqkv_p + 2 * HD, BD, HD, HD, 0, vp, nullptr, nullptr, nullptr);

  // attention (rd/pd reused as attention outputs)
  attn_kernel<<<dim3(NB, NHh), 256, 0, stream>>>(qr, kr, vr, npi, rd);
  attn_kernel<<<dim3(NB, NHh), 256, 0, stream>>>(qp, kp, vp, nri, pd);

  // output projection + gather residual: att[node] = nf[node] + attn @ Wo^T + bo
  gemm(rd, wor, bo_r, BD, HD, HD, 1, attb, att, nullptr, nidr);
  gemm(pd, wop, bo_p, BD, HD, HD, 1, attb, att, nullptr, nidp);

  // FFN
  gemm(attb, w1b, b1, Nn, HD, 2 * HD, 2, h1, nullptr, nullptr, nullptr);
  gemm(h1, w2b, b2, Nn, 2 * HD, HD, 3, nullptr, (float*)d_out, att, nullptr);
}
